// TransformerGraphAutoencoder_2714419331087
// MI455X (gfx1250) — matrix operand order repaired and checked by reading, not compiled, not run on hardware
//
// CDNA5 / gfx1250 implementation of the TransformerGraphAutoencoder forward.
// Big GEMMs: v_wmma_f32_16x16x32_f16, f32 accumulate, with the shared B panel
// staged into LDS by the Tensor Data Mover (tensor_load_to_lds, TENSORcnt)
// and double-buffered; A streams from global with WGP-scope prefetch.
#include <hip/hip_runtime.h>
#include <hip/hip_bf16.h>
#include <stdint.h>
#include <stddef.h>

#define BATCH  4
#define NPTS   1024
#define DIM    1024
#define FLAT   512
#define HEADS  8
#define DH     128
#define KNNK   16
#define NCLS   16
#define MTOK   (BATCH*NPTS)
#define BH     (BATCH*HEADS)
#define NEGV   (-1.0e9f)
#define EPSV   (1e-5f)
#define BN_S   (0.9999950000374997f)   /* 1/sqrt(1+1e-5) */
#define FILM_KC 0.5f

typedef __attribute__((ext_vector_type(16))) _Float16 v16h;
typedef __attribute__((ext_vector_type(8)))  _Float16 v8h;
typedef __attribute__((ext_vector_type(8)))  float    v8f;
typedef __attribute__((ext_vector_type(4)))  unsigned int u32x4;
typedef __attribute__((ext_vector_type(8)))  int      i32x8;
typedef __attribute__((ext_vector_type(4)))  int      i32x4;

// ---------------------------------------------------------------- helpers
__device__ __forceinline__ float wave_sum(float v) {
#pragma unroll
  for (int m = 16; m >= 1; m >>= 1) v += __shfl_xor(v, m, 32);
  return v;
}
__device__ __forceinline__ float wave_max(float v) {
#pragma unroll
  for (int m = 16; m >= 1; m >>= 1) v = fmaxf(v, __shfl_xor(v, m, 32));
  return v;
}

// A-fragment (16x32 f16): lane = row (lane&15); the lane's 16 halfs are
// contiguous K[hi*16 .. hi*16+15]  (hi = lane>>4), matching ldB.
__device__ __forceinline__ v16h ldA(const _Float16* p, int hi) {
  return *(const v16h*)(p + hi * 16);
}
// B-fragment (32x16 f16): lane = column (row of W = B^T), contiguous 16 K-halfs
// at K offset hi*16.
__device__ __forceinline__ v16h ldB(const _Float16* p, int hi) {
  return *(const v16h*)(p + hi * 16);
}
__device__ __forceinline__ v8f wmma_f16(v16h a, v16h b, v8f c) {
  return __builtin_amdgcn_wmma_f32_16x16x32_f16(false, a, false, b, (short)0, c,
                                                false, false);
}

// Tensor Data Mover: 2D tile load (rows x kc f16) global -> LDS.
// Pads 16B after each row (64-DWORD interval for kc=128, 32-DWORD for kc=64)
// so LDS row stride is (kc+8) halves -> conflict-free ds_load_b128.
__device__ __forceinline__ void tdm_load_b_tile(unsigned ldsOff, const _Float16* g,
                                                int kc, int rows, long stride_elems) {
  unsigned long long ga = (unsigned long long)(uintptr_t)g;
  u32x4 g0;
  g0.x = 1u;                                   // count=1 (valid user descriptor)
  g0.y = ldsOff;                               // LDS byte address
  g0.z = (unsigned)ga;                         // global_addr[31:0]
  g0.w = (unsigned)((ga >> 32) & 0x1FFFFFFull) | (2u << 30);  // addr[56:32]|type=2
  unsigned padItv = (kc == 128) ? 5u : 4u;     // 64 / 32 DWORDs between pads
  unsigned w0 = (1u << 16)                     // data_size = 2 bytes
              | (1u << 20)                     // pad_enable
              | (padItv << 22)                 // pad_interval
              | (3u << 25);                    // pad_amount = 4 DWORDs (16B)
  unsigned td0 = (unsigned)kc, td1 = (unsigned)rows;
  unsigned long long s0 = (unsigned long long)stride_elems;
  i32x8 g1;
  g1[0] = (int)w0;
  g1[1] = (int)((td0 & 0xFFFFu) << 16);                       // tensor_dim0 lo
  g1[2] = (int)((td0 >> 16) | ((td1 & 0xFFFFu) << 16));       // dim0 hi | dim1 lo
  g1[3] = (int)((td1 >> 16) | (((unsigned)kc & 0xFFFFu) << 16)); // dim1 hi|tile0
  g1[4] = (int)((unsigned)rows & 0xFFFFu);                    // tile_dim1, tile2=0
  g1[5] = (int)(unsigned)(s0 & 0xFFFFFFFFull);                // dim0_stride lo
  g1[6] = (int)(unsigned)((s0 >> 32) & 0xFFFFull);            // dim0_stride hi
  g1[7] = 0;
  i32x4 gz = {0, 0, 0, 0};                                    // dims 2..4 unused
  i32x8 gz8 = {0, 0, 0, 0, 0, 0, 0, 0};
  __builtin_amdgcn_tensor_load_to_lds(g0, g1, gz, gz, gz8, 0);
}

// ---------------------------------------------------------------- generic GEMM
// C[M x Nc] = alpha * A[M x K] * W[Nc x K]^T  (+bias)(*bn)(relu), f32/f16 store.
// Block = 256 threads (8 waves); wave = 16 rows x 64 cols (4 accumulators).
// B panel (shared by all 8 waves) is TDM-staged into LDS, double buffered.
#define GF_RELU    1
#define GF_ST16    2
#define GF_ST32    4
#define GF_BIASROW 8
#define GF_BN      16

#define BKSLAB   128
#define LDSTRMAX 136               /* 128 + 8 pad halves */
#define BUFELEMS (64 * LDSTRMAX)   /* per ping-pong buffer */
#define BUFBYTES (BUFELEMS * 2)

__global__ __launch_bounds__(256)
void wmma_gemm_kernel(const _Float16* __restrict__ A, int lda,
                      const _Float16* __restrict__ W, int ldb,
                      float* __restrict__ Cf, _Float16* __restrict__ Ch, int ldc,
                      const float* __restrict__ bias,
                      const float* __restrict__ scale,
                      const float* __restrict__ shift,
                      float alpha, int Kd, int flags) {
  __shared__ _Float16 Bsh[2 * BUFELEMS];   // first (only) LDS object -> offset 0
  const int wave = threadIdx.x >> 5, lane = threadIdx.x & 31;
  const int lr = lane & 15, hi = lane >> 4;
  const int m0 = blockIdx.x * 128 + wave * 16;
  const int n0 = blockIdx.y * 64;
  const _Float16* aP = A + (size_t)(m0 + lr) * lda;
  const _Float16* bBase = W + (size_t)n0 * ldb;
  const int nslab = (Kd + BKSLAB - 1) / BKSLAB;
  const bool issuer = (threadIdx.x < 32);   // wave 0 drives the TDM

  // The TDM builtin writes LDS as a side effect clang does not model; give the
  // optimizer a reaching may-def of Bsh (laundered pointer, never-true guard)
  // so the ds_load fragments below cannot be folded to undef/zero.
  {
    _Float16* p0 = &Bsh[0];
    asm volatile("" : "+v"(p0));
    if (Kd < 0) *p0 = (_Float16)0;
  }

  if (issuer) {
    int kc0 = (Kd < BKSLAB) ? Kd : BKSLAB;
    tdm_load_b_tile(0u, bBase, kc0, 64, (long)ldb);
    __builtin_amdgcn_s_wait_tensorcnt((short)0);
    asm volatile("" ::: "memory");
  }
  __syncthreads();

  v8f a0 = {}, a1 = {}, a2 = {}, a3 = {};
  for (int s = 0; s < nslab; s++) {
    const int kc = (Kd - s * BKSLAB < BKSLAB) ? (Kd - s * BKSLAB) : BKSLAB;
    if (issuer && s + 1 < nslab) {
      int kcn = (Kd - (s + 1) * BKSLAB < BKSLAB) ? (Kd - (s + 1) * BKSLAB) : BKSLAB;
      tdm_load_b_tile((unsigned)(((s + 1) & 1) * BUFBYTES),
                      bBase + (size_t)(s + 1) * BKSLAB, kcn, 64, (long)ldb);
    }
    const _Float16* aS = aP + (size_t)s * BKSLAB;
    if (s + 1 < nslab)                       // WGP-scope prefetch of next A slab
      __builtin_prefetch(aP + (size_t)(s + 1) * BKSLAB, 0, 3);
    const int stride = kc + 8;               // LDS row stride (halves)
    const _Float16* bs = &Bsh[(s & 1) * BUFELEMS] + (size_t)lr * stride;
    v16h af = ldA(aS, hi);                   // one-step software pipeline on A
    for (int ks = 0; ks < kc; ks += 32) {
      v16h afn = af;
      if (ks + 32 < kc) afn = ldA(aS + ks + 32, hi);
      const _Float16* bRow = bs + ks;
      a0 = wmma_f16(af, ldB(bRow, hi), a0);
      a1 = wmma_f16(af, ldB(bRow + 16 * stride, hi), a1);
      a2 = wmma_f16(af, ldB(bRow + 32 * stride, hi), a2);
      a3 = wmma_f16(af, ldB(bRow + 48 * stride, hi), a3);
      af = afn;
    }
    if (issuer && s + 1 < nslab) {
      __builtin_amdgcn_s_wait_tensorcnt((short)0);
      asm volatile("" ::: "memory");
    }
    __syncthreads();                         // buffer handoff
  }

  v8f accs[4] = {a0, a1, a2, a3};
#pragma unroll
  for (int j = 0; j < 4; j++) {
    int c = n0 + j * 16 + lr;
#pragma unroll
    for (int i = 0; i < 8; i++) {
      int r = m0 + i + 8 * hi;
      float v = accs[j][i] * alpha;
      if (bias) v += (flags & GF_BIASROW) ? bias[r] : bias[c];
      if (flags & GF_BN) v = v * (scale[c] * BN_S) + shift[c];
      if (flags & GF_RELU) v = fmaxf(v, 0.0f);
      size_t o = (size_t)r * ldc + c;
      if (flags & GF_ST32) Cf[o] = v;
      if (flags & GF_ST16) Ch[o] = (_Float16)v;
    }
  }
}

// ---------------------------------------------------------------- attention
__global__ __launch_bounds__(256)
void attn_scores_kernel(const _Float16* __restrict__ Q,
                        const _Float16* __restrict__ Kmat,
                        _Float16* __restrict__ S, float alpha) {
  const int wave = threadIdx.x >> 5, lane = threadIdx.x & 31;
  const int lr = lane & 15, hi = lane >> 4;
  const int z = blockIdx.z, b = z / HEADS, h = z % HEADS;
  const int q0 = blockIdx.x * 16;
  const int k0 = blockIdx.y * 128 + wave * 16;
  const _Float16* aP = Q    + ((size_t)b * NPTS + q0 + lr) * DIM + h * DH;
  const _Float16* bP = Kmat + ((size_t)b * NPTS + k0 + lr) * DIM + h * DH;
  v8f acc = {};
#pragma unroll
  for (int kk = 0; kk < DH; kk += 32)
    acc = wmma_f16(ldA(aP + kk, hi), ldB(bP + kk, hi), acc);
  _Float16* out = S + (size_t)z * NPTS * NPTS;
#pragma unroll
  for (int i = 0; i < 8; i++) {
    int r = q0 + i + 8 * hi, c = k0 + lr;
    out[(size_t)r * NPTS + c] = (_Float16)(acc[i] * alpha);
  }
}

// O[b, q, h*DH+d] = sum_k P[z,q,k] * Vt[h*DH+d, b*NPTS+k]
__global__ __launch_bounds__(256)
void attn_av_kernel(const _Float16* __restrict__ S,
                    const _Float16* __restrict__ Vt,
                    float* __restrict__ O) {
  const int wave = threadIdx.x >> 5, lane = threadIdx.x & 31;
  const int lr = lane & 15, hi = lane >> 4;
  const int z = blockIdx.z, b = z / HEADS, h = z % HEADS;
  const int q0 = blockIdx.x * 16;
  const int d0 = wave * 16;
  const _Float16* aP = S + (size_t)z * NPTS * NPTS + (size_t)(q0 + lr) * NPTS;
  const _Float16* bP = Vt + (size_t)(h * DH + d0 + lr) * MTOK + (size_t)b * NPTS;
  v8f acc = {};
  for (int kk = 0; kk < NPTS; kk += 32) {
    if (kk + 64 < NPTS) {
      __builtin_prefetch(aP + kk + 64, 0, 3);
      __builtin_prefetch(bP + kk + 64, 0, 3);
    }
    acc = wmma_f16(ldA(aP + kk, hi), ldB(bP + kk, hi), acc);
  }
#pragma unroll
  for (int i = 0; i < 8; i++) {
    int r = q0 + i + 8 * hi, c = h * DH + d0 + lr;
    O[((size_t)b * NPTS + r) * DIM + c] = acc[i];
  }
}

// softmax over rows of S (BH*NPTS rows of NPTS), optional key-padding mask
__global__ __launch_bounds__(256)
void softmax_kernel(_Float16* __restrict__ S, const int* __restrict__ mask) {
  const size_t row = (size_t)blockIdx.x * 8 + (threadIdx.x >> 5);
  const int lane = threadIdx.x & 31;
  const int b = (int)(row / ((size_t)HEADS * NPTS));
  _Float16* sr = S + row * NPTS;
  float vals[NPTS / 32];
  float mx = -3.0e38f;
#pragma unroll
  for (int j = 0; j < NPTS / 32; j++) {
    int c = lane + j * 32;
    float v = (float)sr[c];
    if (mask && mask[b * NPTS + c] == 0) v = NEGV;
    vals[j] = v;
    mx = fmaxf(mx, v);
  }
  mx = wave_max(mx);
  float s = 0.0f;
#pragma unroll
  for (int j = 0; j < NPTS / 32; j++) {
    float e = __expf(vals[j] - mx);
    vals[j] = e; s += e;
  }
  s = wave_sum(s);
  float inv = 1.0f / s;
#pragma unroll
  for (int j = 0; j < NPTS / 32; j++)
    sr[lane + j * 32] = (_Float16)(vals[j] * inv);
}

// Out = LayerNorm(X + Y) * g + b   (one wave per row of DIM)
__global__ __launch_bounds__(256)
void add_ln_kernel(const float* __restrict__ X, const float* __restrict__ Y,
                   float* __restrict__ O, const float* __restrict__ g,
                   const float* __restrict__ bb) {
  const size_t row = (size_t)blockIdx.x * 8 + (threadIdx.x >> 5);
  const int lane = threadIdx.x & 31;
  const float* xr = X + row * DIM;
  const float* yr = Y + row * DIM;
  float vals[DIM / 32];
  float s = 0.0f;
#pragma unroll
  for (int j = 0; j < DIM / 32; j++) {
    float v = xr[lane + j * 32] + yr[lane + j * 32];
    vals[j] = v; s += v;
  }
  s = wave_sum(s);
  float mean = s * (1.0f / DIM);
  float q = 0.0f;
#pragma unroll
  for (int j = 0; j < DIM / 32; j++) { float d = vals[j] - mean; q += d * d; }
  q = wave_sum(q);
  float inv = rsqrtf(q * (1.0f / DIM) + EPSV);
  float* o = O + row * DIM;
#pragma unroll
  for (int j = 0; j < DIM / 32; j++) {
    int c = lane + j * 32;
    o[c] = (vals[j] - mean) * inv * g[c] + bb[c];
  }
}

// ---------------------------------------------------------------- point path
__global__ void knn_kernel(const float* __restrict__ X, const int* __restrict__ mask,
                           int* __restrict__ idx) {
  int t = blockIdx.x * blockDim.x + threadIdx.x;
  if (t >= BATCH * NPTS) return;
  int b = t / NPTS, n = t % NPTS;
  const float* xb = X + (size_t)b * 3 * NPTS;
  float px = xb[n], py = xb[NPTS + n], pz = xb[2 * NPTS + n];
  float bd[KNNK]; int bi[KNNK];
  for (int i = 0; i < KNNK; i++) { bd[i] = 3.0e38f; bi[i] = 0; }
  for (int m = 0; m < NPTS; m++) {
    float dx = px - xb[m], dy = py - xb[NPTS + m], dz = pz - xb[2 * NPTS + m];
    float d2 = dx * dx + dy * dy + dz * dz;
    if (mask[b * NPTS + m] == 0) d2 = 1.0e30f;
    if (d2 < bd[KNNK - 1]) {
      int j = KNNK - 1;
      while (j > 0 && bd[j - 1] > d2) { bd[j] = bd[j - 1]; bi[j] = bi[j - 1]; j--; }
      bd[j] = d2; bi[j] = m;
    }
  }
  for (int i = 0; i < KNNK; i++) idx[(size_t)t * KNNK + i] = bi[i];
}

__global__ void cov_mlp1_kernel(const float* __restrict__ X, const int* __restrict__ idx,
                                const float* W1, const float* b1, const float* g1, const float* be1,
                                const float* W2, const float* b2, const float* g2, const float* be2,
                                float* __restrict__ hf) {
  int t = blockIdx.x * blockDim.x + threadIdx.x;
  if (t >= BATCH * NPTS) return;
  int b = t / NPTS, n = t % NPTS;
  const float* xb = X + (size_t)b * 3 * NPTS;
  float nb[KNNK][3], mu[3] = {0, 0, 0};
  for (int k = 0; k < KNNK; k++) {
    int m = idx[(size_t)t * KNNK + k];
    nb[k][0] = xb[m]; nb[k][1] = xb[NPTS + m]; nb[k][2] = xb[2 * NPTS + m];
    mu[0] += nb[k][0]; mu[1] += nb[k][1]; mu[2] += nb[k][2];
  }
  mu[0] *= (1.0f / KNNK); mu[1] *= (1.0f / KNNK); mu[2] *= (1.0f / KNNK);
  float cov[9] = {0};
  for (int k = 0; k < KNNK; k++) {
    float c0 = nb[k][0] - mu[0], c1 = nb[k][1] - mu[1], c2 = nb[k][2] - mu[2];
    cov[0] += c0 * c0; cov[1] += c0 * c1; cov[2] += c0 * c2;
    cov[3] += c1 * c0; cov[4] += c1 * c1; cov[5] += c1 * c2;
    cov[6] += c2 * c0; cov[7] += c2 * c1; cov[8] += c2 * c2;
  }
  float f12[12] = {xb[n], xb[NPTS + n], xb[2 * NPTS + n],
                   cov[0], cov[1], cov[2], cov[3], cov[4], cov[5], cov[6], cov[7], cov[8]};
  float h1[64];
  for (int o = 0; o < 64; o++) {
    float a = b1[o];
    for (int c = 0; c < 12; c++) a += W1[o * 12 + c] * f12[c];
    a = a * (g1[o] * BN_S) + be1[o];
    h1[o] = fmaxf(a, 0.0f);
  }
  float* out = hf + (size_t)t * 64;
  for (int o = 0; o < 64; o++) {
    float a = b2[o];
    for (int c = 0; c < 64; c++) a += W2[o * 64 + c] * h1[c];
    a = a * (g2[o] * BN_S) + be2[o];
    out[o] = fmaxf(a, 0.0f);
  }
}

__global__ void nbr_mean_kernel(const float* __restrict__ hf, const int* __restrict__ idx,
                                _Float16* __restrict__ pm16) {
  int t = blockIdx.x * blockDim.x + threadIdx.x;
  if (t >= BATCH * NPTS) return;
  int b = t / NPTS;
  float acc[64];
  for (int o = 0; o < 64; o++) acc[o] = 0.0f;
  for (int k = 0; k < KNNK; k++) {
    int m = idx[(size_t)t * KNNK + k];
    const float* r = hf + ((size_t)b * NPTS + m) * 64;
    for (int o = 0; o < 64; o++) acc[o] += r[o];
  }
  _Float16* o16 = pm16 + (size_t)t * 64;
  for (int o = 0; o < 64; o++) o16[o] = (_Float16)(acc[o] * (1.0f / KNNK));
}

// ---------------------------------------------------------------- misc small
__global__ void f32_to_f16_kernel(const float* __restrict__ s, _Float16* __restrict__ d, int n) {
  int i = blockIdx.x * blockDim.x + threadIdx.x;
  if (i < n) d[i] = (_Float16)s[i];
}

__global__ void pe_kernel(float* __restrict__ pe, _Float16* __restrict__ pe16) {
  int i = blockIdx.x * blockDim.x + threadIdx.x;
  if (i >= NPTS * DIM) return;
  int n = i / DIM, d = i % DIM;
  int ii = d >> 1;
  float ang = (float)n * __expf(-(2.0f * (float)ii / (float)DIM) * 9.210340371976184f);
  float v = (d & 1) ? __cosf(ang) : __sinf(ang);
  pe[i] = v; pe16[i] = (_Float16)v;
}

// out[b,n,d] = G[n,d]*src[b*sB + n*sN + d] + Bt[n,d]
__global__ void film_kernel(const float* __restrict__ G, const float* __restrict__ Bt,
                            const float* __restrict__ src, float* __restrict__ out,
                            int sB, int sN) {
  int i = blockIdx.x * blockDim.x + threadIdx.x;
  if (i >= MTOK * DIM) return;
  int d = i % DIM;
  int nd = i / DIM;
  int n = nd % NPTS, b = nd / NPTS;
  out[i] = G[(size_t)n * DIM + d] * src[(size_t)b * sB + (size_t)n * sN + d] +
           Bt[(size_t)n * DIM + d];
}

__global__ void maxpool_kernel(const float* __restrict__ X, const int* __restrict__ mask,
                               float* __restrict__ pooled) {
  int t = blockIdx.x * blockDim.x + threadIdx.x;
  if (t >= BATCH * DIM) return;
  int b = t / DIM, d = t % DIM;
  float m = NEGV;
  for (int n = 0; n < NPTS; n++)
    if (mask[b * NPTS + n]) m = fmaxf(m, X[((size_t)b * NPTS + n) * DIM + d]);
  pooled[t] = m;
}

__global__ void dense_bn_relu_kernel(const float* __restrict__ W, const float* __restrict__ bias,
                                     const float* __restrict__ g, const float* __restrict__ be,
                                     const float* __restrict__ in, float* __restrict__ out,
                                     int inF, int outD) {
  int t = blockIdx.x * blockDim.x + threadIdx.x;
  if (t >= BATCH * outD) return;
  int b = t / outD, d = t % outD;
  float a = bias[d];
  const float* ir = in + (size_t)b * inF;
  for (int f = 0; f < inF; f++) a += W[(size_t)d * inF + f] * ir[f];
  a = a * (g[d] * BN_S) + be[d];
  out[t] = fmaxf(a, 0.0f);
}

__global__ void dense_kernel(const float* __restrict__ W, const float* __restrict__ bias,
                             const float* __restrict__ in, float* __restrict__ out,
                             int inF, int outD) {
  int t = blockIdx.x * blockDim.x + threadIdx.x;
  if (t >= BATCH * outD) return;
  int b = t / outD, d = t % outD;
  float a = bias[d];
  const float* ir = in + (size_t)b * inF;
  for (int f = 0; f < inF; f++) a += W[(size_t)d * inF + f] * ir[f];
  out[t] = a;
}

__global__ void pc_film_kernel(const float* __restrict__ ga, const float* __restrict__ bt,
                               const float* __restrict__ pooled, float* __restrict__ pc) {
  int t = blockIdx.x * blockDim.x + threadIdx.x;
  if (t >= BATCH * DIM) return;
  float p = pooled[t];
  pc[t] = FILM_KC * (ga[t] * p + bt[t]) + (1.0f - FILM_KC) * p;
}

__global__ void znorm_kernel(const float* __restrict__ zpre, float* __restrict__ z,
                             float* __restrict__ zout) {
  int b = blockIdx.x, lane = threadIdx.x;
  float s = 0.0f;
  for (int j = lane; j < FLAT; j += 32) { float v = zpre[b * FLAT + j]; s += v * v; }
  s = wave_sum(s);
  float inv = 1.0f / fmaxf(sqrtf(s), 1e-12f);
  for (int j = lane; j < FLAT; j += 32) {
    float v = zpre[b * FLAT + j] * inv;
    z[b * FLAT + j] = v; zout[b * FLAT + j] = v;
  }
}

__global__ void out_final_kernel(const _Float16* __restrict__ h2, const float* __restrict__ W2,
                                 const float* __restrict__ b2, float* __restrict__ out) {
  int t = blockIdx.x * blockDim.x + threadIdx.x;
  if (t >= BATCH * 3 * NPTS) return;
  int b = t / (3 * NPTS), rem = t % (3 * NPTS);
  int f = rem / NPTS, n = rem % NPTS;
  const _Float16* hr = h2 + ((size_t)b * NPTS + n) * FLAT;
  float a = b2[f];
  for (int j = 0; j < FLAT; j++) a += W2[f * FLAT + j] * (float)hr[j];
  out[t] = a;
}

// ---------------------------------------------------------------- host
struct TEP {
  const float *Wq, *Wk, *Wv, *Wo, *bq, *bk, *bv, *bo, *ln1g, *ln1b,
              *W1, *b1, *W2, *b2, *ln2g, *ln2b;
};
struct TE16 { _Float16 *Wq, *Wk, *Wv, *Wo, *W1, *W2; };

extern "C" void kernel_launch(void* const* d_in, const int* in_sizes, int n_in,
                              void* d_out, int out_size, void* d_ws, size_t ws_size,
                              hipStream_t stream) {
  (void)in_sizes; (void)n_in; (void)out_size; (void)ws_size;
#define FIN(i) ((const float*)d_in[i])
  const float* X    = FIN(0);
  const float* CLSS = FIN(1);
  const int*   MASK = (const int*)d_in[2];
  const float *mlp1_W1 = FIN(3), *mlp1_b1 = FIN(4), *mlp1_g1 = FIN(5), *mlp1_be1 = FIN(6);
  const float *mlp1_W2 = FIN(7), *mlp1_b2 = FIN(8), *mlp1_g2 = FIN(9), *mlp1_be2 = FIN(10);
  const float *gc_W = FIN(11), *gc_b = FIN(12);
  const float *pee_Wg = FIN(13), *pee_bg = FIN(14), *pee_Wb = FIN(15), *pee_bb = FIN(16);
  auto loadTE = [&](int base) {
    TEP t;
    t.Wq = FIN(base + 0);  t.Wk = FIN(base + 1);  t.Wv = FIN(base + 2);  t.Wo = FIN(base + 3);
    t.bq = FIN(base + 4);  t.bk = FIN(base + 5);  t.bv = FIN(base + 6);  t.bo = FIN(base + 7);
    t.ln1g = FIN(base + 8); t.ln1b = FIN(base + 9);
    t.W1 = FIN(base + 10); t.b1 = FIN(base + 11); t.W2 = FIN(base + 12); t.b2 = FIN(base + 13);
    t.ln2g = FIN(base + 14); t.ln2b = FIN(base + 15);
    return t;
  };
  TEP P[4] = { loadTE(17), loadTE(33), loadTE(69), loadTE(85) }; // enc0 enc1 dec0 dec1
  const float *mc_W = FIN(49), *mc_b = FIN(50), *mc_g = FIN(51), *mc_be = FIN(52);
  const float *ga_W = FIN(53), *ga_b = FIN(54), *ga_g = FIN(55), *ga_be = FIN(56);
  const float *bt_W = FIN(57), *bt_b = FIN(58), *bt_g = FIN(59), *bt_be = FIN(60);
  const float *fce_W = FIN(61), *fce_b = FIN(62), *fcd_W = FIN(63), *fcd_b = FIN(64);
  const float *ped_Wg = FIN(65), *ped_bg = FIN(66), *ped_Wb = FIN(67), *ped_bb = FIN(68);
  const float *m2_W1 = FIN(101), *m2_b1 = FIN(102), *m2_g1 = FIN(103), *m2_be1 = FIN(104);
  const float *m2_W2 = FIN(105), *m2_b2 = FIN(106);

  // ---- bump allocator over workspace (all buffers 256B aligned)
  uintptr_t cur = ((uintptr_t)d_ws + 255) & ~(uintptr_t)255;
  auto alloc = [&](size_t bytes) -> void* {
    void* p = (void*)cur;
    cur = (cur + bytes + 255) & ~(uintptr_t)255;
    return p;
  };
  float*    PEf  = (float*)alloc((size_t)NPTS * DIM * 4);
  _Float16* PE16 = (_Float16*)alloc((size_t)NPTS * DIM * 2);
  TE16 W16[4];
  for (int l = 0; l < 4; l++) {
    W16[l].Wq = (_Float16*)alloc((size_t)DIM * DIM * 2);
    W16[l].Wk = (_Float16*)alloc((size_t)DIM * DIM * 2);
    W16[l].Wv = (_Float16*)alloc((size_t)DIM * DIM * 2);
    W16[l].Wo = (_Float16*)alloc((size_t)DIM * DIM * 2);
    W16[l].W1 = (_Float16*)alloc((size_t)DIM * DIM * 2);
    W16[l].W2 = (_Float16*)alloc((size_t)DIM * DIM * 2);
  }
  _Float16* peeWg16 = (_Float16*)alloc((size_t)DIM * DIM * 2);
  _Float16* peeWb16 = (_Float16*)alloc((size_t)DIM * DIM * 2);
  _Float16* pedWg16 = (_Float16*)alloc((size_t)DIM * DIM * 2);
  _Float16* pedWb16 = (_Float16*)alloc((size_t)DIM * DIM * 2);
  _Float16* gc16    = (_Float16*)alloc((size_t)DIM * 64 * 2);
  _Float16* m2W116  = (_Float16*)alloc((size_t)FLAT * DIM * 2);
  float* Gge = (float*)alloc((size_t)NPTS * DIM * 4);
  float* Gbe = (float*)alloc((size_t)NPTS * DIM * 4);
  float* Ggd = (float*)alloc((size_t)NPTS * DIM * 4);
  float* Gbd = (float*)alloc((size_t)NPTS * DIM * 4);
  int*      idx  = (int*)alloc((size_t)MTOK * KNNK * 4);
  float*    hf   = (float*)alloc((size_t)MTOK * 64 * 4);
  _Float16* pm16 = (_Float16*)alloc((size_t)MTOK * 64 * 2);
  float*    xa   = (float*)alloc((size_t)MTOK * DIM * 4);
  float*    xb   = (float*)alloc((size_t)MTOK * DIM * 4);
  _Float16* x16  = (_Float16*)alloc((size_t)MTOK * DIM * 2);
  _Float16* q16  = (_Float16*)alloc((size_t)MTOK * DIM * 2);
  _Float16* k16  = (_Float16*)alloc((size_t)MTOK * DIM * 2);
  _Float16* vt16 = (_Float16*)alloc((size_t)DIM * MTOK * 2);
  _Float16* S16  = (_Float16*)alloc((size_t)BH * NPTS * NPTS * 2);
  float* pooled = (float*)alloc((size_t)BATCH * DIM * 4);
  float* ce     = (float*)alloc((size_t)BATCH * FLAT * 4);
  float* gaB    = (float*)alloc((size_t)BATCH * DIM * 4);
  float* btB    = (float*)alloc((size_t)BATCH * DIM * 4);
  float* pc     = (float*)alloc((size_t)BATCH * DIM * 4);
  float* zpre   = (float*)alloc((size_t)BATCH * FLAT * 4);
  float* zb     = (float*)alloc((size_t)BATCH * FLAT * 4);
  float* d0v    = (float*)alloc((size_t)BATCH * DIM * 4);
  float* OUT = (float*)d_out;

  auto cvt = [&](const float* s, _Float16* d, size_t n) {
    f32_to_f16_kernel<<<dim3((unsigned)((n + 255) / 256)), 256, 0, stream>>>(s, d, (int)n);
  };
  auto gemm = [&](const _Float16* A, int lda, const _Float16* W, int ldb,
                  float* Cf, _Float16* Ch, int ldc, const float* bias, float alpha,
                  int M, int Nc, int Kd, int flags,
                  const float* scale = nullptr, const float* shift = nullptr) {
    wmma_gemm_kernel<<<dim3(M / 128, Nc / 64, 1), 256, 0, stream>>>(
        A, lda, W, ldb, Cf, Ch, ldc, bias, scale, shift, alpha, Kd, flags);
  };
  auto addln = [&](const float* Xi, const float* Yi, float* Oi, const float* g, const float* b) {
    add_ln_kernel<<<dim3(MTOK / 8), 256, 0, stream>>>(Xi, Yi, Oi, g, b);
  };

  // ---- constants / weight precision conversion
  pe_kernel<<<dim3(NPTS * DIM / 256), 256, 0, stream>>>(PEf, PE16);
  for (int l = 0; l < 4; l++) {
    cvt(P[l].Wq, W16[l].Wq, (size_t)DIM * DIM);
    cvt(P[l].Wk, W16[l].Wk, (size_t)DIM * DIM);
    cvt(P[l].Wv, W16[l].Wv, (size_t)DIM * DIM);
    cvt(P[l].Wo, W16[l].Wo, (size_t)DIM * DIM);
    cvt(P[l].W1, W16[l].W1, (size_t)DIM * DIM);
    cvt(P[l].W2, W16[l].W2, (size_t)DIM * DIM);
  }
  cvt(pee_Wg, peeWg16, (size_t)DIM * DIM);
  cvt(pee_Wb, peeWb16, (size_t)DIM * DIM);
  cvt(ped_Wg, pedWg16, (size_t)DIM * DIM);
  cvt(ped_Wb, pedWb16, (size_t)DIM * DIM);
  cvt(gc_W, gc16, (size_t)DIM * 64);
  cvt(m2_W1, m2W116, (size_t)FLAT * DIM);

  // FiLM gamma/beta tables: G = PE @ W^T + b   (N x D)
  gemm(PE16, DIM, peeWg16, DIM, Gge, nullptr, DIM, pee_bg, 1.0f, NPTS, DIM, DIM, GF_ST32);
  gemm(PE16, DIM, peeWb16, DIM, Gbe, nullptr, DIM, pee_bb, 1.0f, NPTS, DIM, DIM, GF_ST32);
  gemm(PE16, DIM, pedWg16, DIM, Ggd, nullptr, DIM, ped_bg, 1.0f, NPTS, DIM, DIM, GF_ST32);
  gemm(PE16, DIM, pedWb16, DIM, Gbd, nullptr, DIM, ped_bb, 1.0f, NPTS, DIM, DIM, GF_ST32);

  // ---- point features
  knn_kernel<<<dim3(MTOK / 256), 256, 0, stream>>>(X, MASK, idx);
  cov_mlp1_kernel<<<dim3(MTOK / 256), 256, 0, stream>>>(
      X, idx, mlp1_W1, mlp1_b1, mlp1_g1, mlp1_be1, mlp1_W2, mlp1_b2, mlp1_g2, mlp1_be2, hf);
  nbr_mean_kernel<<<dim3(MTOK / 256), 256, 0, stream>>>(hf, idx, pm16);
  // g = pm @ gc_W^T + gc_b  -> xb (MTOK x DIM)
  gemm(pm16, 64, gc16, 64, xb, nullptr, DIM, gc_b, 1.0f, MTOK, DIM, 64, GF_ST32);
  // encoder FiLM
  film_kernel<<<dim3(MTOK * DIM / 256), 256, 0, stream>>>(Gge, Gbe, xb, xa, NPTS * DIM, DIM);

  // ---- transformer layer
  auto run_layer = [&](const TEP& p, const TE16& w, const int* maskPtr) {
    cvt(xa, x16, (size_t)MTOK * DIM);
    gemm(x16, DIM, w.Wq, DIM, nullptr, q16, DIM, p.bq, 1.0f, MTOK, DIM, DIM, GF_ST16);
    gemm(x16, DIM, w.Wk, DIM, nullptr, k16, DIM, p.bk, 1.0f, MTOK, DIM, DIM, GF_ST16);
    // role-swapped: Vt[d, token] = Wv[d,:] . x[token,:] + bv[d]
    gemm(w.Wv, DIM, x16, DIM, nullptr, vt16, MTOK, p.bv, 1.0f, DIM, MTOK, DIM,
         GF_ST16 | GF_BIASROW);
    attn_scores_kernel<<<dim3(NPTS / 16, NPTS / 128, BH), 256, 0, stream>>>(
        q16, k16, S16, 0.08838834764831845f);
    softmax_kernel<<<dim3(BH * NPTS / 8), 256, 0, stream>>>(S16, maskPtr);
    attn_av_kernel<<<dim3(NPTS / 16, 1, BH), 256, 0, stream>>>(S16, vt16, xb);
    cvt(xb, x16, (size_t)MTOK * DIM);
    gemm(x16, DIM, w.Wo, DIM, xb, nullptr, DIM, p.bo, 1.0f, MTOK, DIM, DIM, GF_ST32);
    addln(xa, xb, xa, p.ln1g, p.ln1b);
    cvt(xa, x16, (size_t)MTOK * DIM);
    gemm(x16, DIM, w.W1, DIM, nullptr, q16, DIM, p.b1, 1.0f, MTOK, DIM, DIM,
         GF_ST16 | GF_RELU);
    gemm(q16, DIM, w.W2, DIM, xb, nullptr, DIM, p.b2, 1.0f, MTOK, DIM, DIM, GF_ST32);
    addln(xa, xb, xa, p.ln2g, p.ln2b);
  };

  run_layer(P[0], W16[0], MASK);
  run_layer(P[1], W16[1], MASK);

  // ---- pooled latent + class FiLM
  maxpool_kernel<<<dim3((BATCH * DIM + 255) / 256), 256, 0, stream>>>(xa, MASK, pooled);
  dense_bn_relu_kernel<<<dim3((BATCH * FLAT + 255) / 256), 256, 0, stream>>>(
      mc_W, mc_b, mc_g, mc_be, CLSS, ce, NCLS, FLAT);
  dense_bn_relu_kernel<<<dim3((BATCH * DIM + 255) / 256), 256, 0, stream>>>(
      ga_W, ga_b, ga_g, ga_be, ce, gaB, FLAT, DIM);
  dense_bn_relu_kernel<<<dim3((BATCH * DIM + 255) / 256), 256, 0, stream>>>(
      bt_W, bt_b, bt_g, bt_be, ce, btB, FLAT, DIM);
  pc_film_kernel<<<dim3((BATCH * DIM + 255) / 256), 256, 0, stream>>>(gaB, btB, pooled, pc);
  dense_kernel<<<dim3((BATCH * FLAT + 255) / 256), 256, 0, stream>>>(
      fce_W, fce_b, pc, zpre, DIM, FLAT);
  znorm_kernel<<<dim3(BATCH), 32, 0, stream>>>(zpre, zb, OUT + BATCH * 3 * NPTS);
  dense_kernel<<<dim3((BATCH * DIM + 255) / 256), 256, 0, stream>>>(
      fcd_W, fcd_b, zb, d0v, FLAT, DIM);

  // ---- decoder
  film_kernel<<<dim3(MTOK * DIM / 256), 256, 0, stream>>>(Ggd, Gbd, d0v, xa, DIM, 0);
  run_layer(P[2], W16[2], nullptr);
  run_layer(P[3], W16[3], nullptr);

  // ---- output head: h2 = relu(bn(x @ m2_W1^T + b1)) (f16), then 3-wide proj
  cvt(xa, x16, (size_t)MTOK * DIM);
  gemm(x16, DIM, m2W116, DIM, nullptr, q16, FLAT, m2_b1, 1.0f, MTOK, FLAT, DIM,
       GF_ST16 | GF_RELU | GF_BN, m2_g1, m2_be1);
  out_final_kernel<<<dim3((BATCH * 3 * NPTS + 255) / 256), 256, 0, stream>>>(
      q16, m2_W2, m2_b2, OUT);
}
